// Nei_mean_15633680958325
// MI455X (gfx1250) — matrix operand order repaired and checked by reading, not compiled, not run on hardware
//
#include <hip/hip_runtime.h>
#include <math.h>

// ---------------------------------------------------------------------------
// Problem constants (from reference): N=4096, M=33, W=7, F=128, REP=128, HYP=128
// ---------------------------------------------------------------------------
#define NB   4096
#define MM   33
#define WW   7
#define FF   128
#define REP  128
#define HYP  128
#define NW   (NB*WW)          // 28672 rows for batched (w,n) GEMMs

typedef unsigned short u16;
typedef __attribute__((ext_vector_type(16))) __bf16 v16bf;
typedef __attribute__((ext_vector_type(8)))  float  v8f;

union V16 { v16bf bf; uint4 q[2]; };

__device__ __forceinline__ u16 f2bf(float f) {
    unsigned u = __float_as_uint(f);
    unsigned r = u + 0x7FFFu + ((u >> 16) & 1u);   // round-to-nearest-even
    return (u16)(r >> 16);
}
__device__ __forceinline__ float sigf(float x) { return 1.0f / (1.0f + __expf(-x)); }

// ---------------------------------------------------------------------------
// fp32 -> bf16 weight conversion
// ---------------------------------------------------------------------------
__global__ void f2bf_kernel(const float* __restrict__ src, u16* __restrict__ dst, int n) {
    int i = blockIdx.x * blockDim.x + threadIdx.x;
    if (i < n) dst[i] = f2bf(src[i]);
}

// ---------------------------------------------------------------------------
// Phase A (bandwidth bound, 484 MB stream): mean over 32 neighbors + target slice.
// Output layout already (w, n, f) so downstream GEMMs are flat (w*n) x f.
// One wave per (n,w) row; lane handles 4 consecutive f.
// ---------------------------------------------------------------------------
__global__ __launch_bounds__(256)
void mean_target_kernel(const float* __restrict__ X, float* __restrict__ xmean,
                        float* __restrict__ tgt) {
    int wid  = (blockIdx.x * blockDim.x + threadIdx.x) >> 5;
    int lane = threadIdx.x & 31;
    if (wid >= NW) return;
    int n = wid / WW, w = wid - n * WW;
    const float* base = X + ((size_t)n * MM * WW + w) * FF;   // (n, m=0, w, :)
    float4 t = *(const float4*)(base + lane * 4);
    float sx = 0.f, sy = 0.f, sz = 0.f, sw = 0.f;
    for (int m = 1; m < MM; ++m) {
        float4 v = *(const float4*)(base + (size_t)m * WW * FF + lane * 4);
        sx += v.x; sy += v.y; sz += v.z; sw += v.w;
    }
    size_t o = ((size_t)w * NB + n) * FF + lane * 4;
    const float inv = 1.0f / 32.0f;
    float4 mv; mv.x = sx * inv; mv.y = sy * inv; mv.z = sz * inv; mv.w = sw * inv;
    *(float4*)(xmean + o) = mv;
    *(float4*)(tgt + o)   = t;
}

// ---------------------------------------------------------------------------
// WMMA GEMM: C[M x N] = act(A[M x K](fp32) @ W[N x K](bf16)^T + bias[N])
// block = 128 threads = 4 waves; block covers 64(M) x 64(N).
// 64 x K A tile staged in LDS as bf16 (<=32 KB of 320 KB/WGP).
// Each wave: one 16-col N tile, FOUR M subtiles -> 4 accumulators; each
// B fragment (2 x b128 global loads, L2-hot weights) is reused by 4 WMMAs.
// A-operand lane layout per CDNA5 ISA 16-bit A 16x32 table (8-elem runs, +16
// interleave); B-operand same 8-elem-run k order as A (column = lane&15).
// ---------------------------------------------------------------------------
template<int ACT>   // 0 = none, 1 = tanh
__global__ __launch_bounds__(128)
void gemm_bias_kernel(const float* __restrict__ A, const u16* __restrict__ Wm,
                      const float* __restrict__ bias, float* __restrict__ C,
                      int M, int N, int K) {
    __shared__ u16 As[64 * 256];
    const int tid = threadIdx.x;
    const int m0  = blockIdx.x * 64;

    // cooperative load+convert of 64 x K fp32 A tile into LDS bf16
    const int kq4 = K >> 2;
    for (int i = tid; i < 64 * kq4; i += 128) {
        int row = i / kq4, kq = i - row * kq4;
        float4 a = *(const float4*)(A + (size_t)(m0 + row) * K + kq * 4);
        unsigned p0 = (unsigned)f2bf(a.x) | ((unsigned)f2bf(a.y) << 16);
        unsigned p1 = (unsigned)f2bf(a.z) | ((unsigned)f2bf(a.w) << 16);
        *(uint2*)&As[row * K + kq * 4] = make_uint2(p0, p1);
    }
    __syncthreads();

    const int wave = tid >> 5, lane = tid & 31;
    const int half = lane >> 4, r16 = lane & 15;
    const int n0 = (blockIdx.y * 4 + wave) * 16;

    v8f acc[4] = {};
    const u16* wrow = Wm + (size_t)(n0 + r16) * K;
    for (int k0 = 0; k0 < K; k0 += 32) {
        V16 bv;
        bv.q[0] = *(const uint4*)(wrow + k0 + half * 8);        // K = k0+h*8 .. +7
        bv.q[1] = *(const uint4*)(wrow + k0 + 16 + half * 8);
        #pragma unroll
        for (int t = 0; t < 4; ++t) {
            const u16* arow = &As[(t * 16 + r16) * K];
            V16 av;
            av.q[0] = *(const uint4*)(arow + k0 + half * 8);      // K = k0+h*8 .. +7
            av.q[1] = *(const uint4*)(arow + k0 + 16 + half * 8); // K = k0+16+h*8 ..
            acc[t] = __builtin_amdgcn_wmma_f32_16x16x32_bf16(
                         false, av.bf, false, bv.bf, (short)0, acc[t], false, false);
        }
    }

    const int col = n0 + r16;
    const float bsv = bias[col];
    #pragma unroll
    for (int t = 0; t < 4; ++t) {
        const int mb = m0 + t * 16 + half * 8;
        #pragma unroll
        for (int r = 0; r < 8; ++r) {
            float v = acc[t][r] + bsv;
            if (ACT == 1) v = tanhf(v);
            C[(size_t)(mb + r) * N + col] = v;
        }
    }
}

// ---------------------------------------------------------------------------
// elementwise kernels
// ---------------------------------------------------------------------------
__global__ void zero_kernel(float* __restrict__ p, int n) {
    int i = blockIdx.x * blockDim.x + threadIdx.x;
    if (i < n) p[i] = 0.0f;
}

// GRU gate fusion: gi/gh are [n,384] (bias already folded by GEMM epilogue)
__global__ __launch_bounds__(256)
void gru_combine_kernel(const float* __restrict__ gi, const float* __restrict__ gh,
                        const float* __restrict__ hin, float* __restrict__ hout,
                        float* __restrict__ hout2) {
    int i = blockIdx.x * blockDim.x + threadIdx.x;      // over NB*REP
    int n = i >> 7, j = i & 127;
    size_t g = (size_t)n * 384 + j;
    float r = sigf(gi[g]       + gh[g]);
    float z = sigf(gi[g + 128] + gh[g + 128]);
    float c = tanhf(gi[g + 256] + r * gh[g + 256]);
    float hn = (1.0f - z) * c + z * hin[i];
    hout[i] = hn;
    if (hout2) hout2[i] = hn;
}

__global__ void concat_kernel(const float* __restrict__ hx, const float* __restrict__ hn,
                              float* __restrict__ out) {
    int i = blockIdx.x * blockDim.x + threadIdx.x;      // over NB*256
    int n = i >> 8, j = i & 255;
    out[i] = (j < 128) ? hx[n * 128 + j] : hn[n * 128 + (j - 128)];
}

// batchnorm training stats: one block per feature (data is L2 resident, 2 MB)
__global__ __launch_bounds__(256)
void bn_stats_kernel(const float* __restrict__ x, float* __restrict__ stats) {
    int f = blockIdx.x, tid = threadIdx.x;
    float s = 0.f, sq = 0.f;
    for (int r = tid; r < NB; r += 256) {
        float v = x[(size_t)r * HYP + f];
        s += v; sq += v * v;
    }
    __shared__ float ss[256], sv[256];
    ss[tid] = s; sv[tid] = sq; __syncthreads();
    for (int o = 128; o > 0; o >>= 1) {
        if (tid < o) { ss[tid] += ss[tid + o]; sv[tid] += sv[tid + o]; }
        __syncthreads();
    }
    if (tid == 0) {
        float m = ss[0] * (1.0f / NB);
        stats[f]       = m;
        stats[128 + f] = sv[0] * (1.0f / NB) - m * m;   // biased var, matches x.var(0)
    }
}

__global__ void bn_apply_relu_kernel(const float* __restrict__ x, const float* __restrict__ stats,
                                     const float* __restrict__ g, const float* __restrict__ b,
                                     float* __restrict__ out) {
    int i = blockIdx.x * blockDim.x + threadIdx.x;      // over NB*HYP
    int j = i & 127;
    float y = g[j] * (x[i] - stats[j]) * rsqrtf(stats[128 + j] + 1e-5f) + b[j];
    out[i] = fmaxf(y, 0.0f);
}

// logits: one wave per row, dot of 128 (N=1 output -> no WMMA needed)
__global__ __launch_bounds__(256)
void logits_kernel(const float* __restrict__ x, const float* __restrict__ w,
                   const float* __restrict__ b, float* __restrict__ out) {
    int wid = (blockIdx.x * blockDim.x + threadIdx.x) >> 5;
    int lane = threadIdx.x & 31;
    if (wid >= NB) return;
    const float* row = x + (size_t)wid * HYP;
    float s = 0.f;
    for (int j = lane; j < HYP; j += 32) s += row[j] * w[j];
    for (int o = 16; o > 0; o >>= 1) s += __shfl_down(s, o, 32);
    if (lane == 0) out[wid] = s + b[0];
}

// BCE terms + sigmoid outputs; deterministic two-stage reduction
__global__ __launch_bounds__(256)
void loss_kernel(const float* __restrict__ logit, const float* __restrict__ Y,
                 float* __restrict__ dout, float* __restrict__ partial) {
    int i = blockIdx.x * 256 + threadIdx.x;
    float term = 0.f;
    if (i < NB) {
        float l = logit[i];
        float t = Y[(size_t)i * MM];                    // target_Y = Y[:,0]
        term = fmaxf(l, 0.0f) - l * t + log1pf(expf(-fabsf(l)));
        dout[1 + i] = 1.0f / (1.0f + expf(-l));
    }
    __shared__ float ss[256];
    ss[threadIdx.x] = term; __syncthreads();
    for (int o = 128; o > 0; o >>= 1) {
        if (threadIdx.x < o) ss[threadIdx.x] += ss[threadIdx.x + o];
        __syncthreads();
    }
    if (threadIdx.x == 0) partial[blockIdx.x] = ss[0];
}

__global__ void loss_final_kernel(const float* __restrict__ partial, float* __restrict__ dout) {
    int lane = threadIdx.x;
    float s = (lane < 16) ? partial[lane] : 0.f;
    for (int o = 16; o > 0; o >>= 1) s += __shfl_down(s, o, 32);
    if (lane == 0) dout[0] = s * (1.0f / NB);
}

// ---------------------------------------------------------------------------
// workspace layout (float elements)
// ---------------------------------------------------------------------------
static constexpr size_t SZ_NWF = (size_t)NW * FF;               // 3,670,016
static constexpr size_t O_XMEAN = 0;
static constexpr size_t O_TGT   = O_XMEAN + SZ_NWF;
static constexpr size_t O_SEQA  = O_TGT   + SZ_NWF;             // [w][n][r]
static constexpr size_t O_SEQB  = O_SEQA  + SZ_NWF;
static constexpr size_t O_HSELF = O_SEQB  + SZ_NWF;
static constexpr size_t O_GI    = O_HSELF + SZ_NWF;             // [w][n][384]
static constexpr size_t O_GH    = O_GI    + (size_t)NW * 384;   // [n][384]
static constexpr size_t O_H     = O_GH    + (size_t)NB * 384;
static constexpr size_t O_CC    = O_H     + (size_t)NB * 128;
static constexpr size_t O_HX    = O_CC    + (size_t)NB * 256;
static constexpr size_t O_Z0    = O_HX    + (size_t)NB * 128;
static constexpr size_t O_Z1    = O_Z0    + (size_t)NB * 128;
static constexpr size_t O_STATS = O_Z1    + (size_t)NB * 128;
static constexpr size_t O_LOG   = O_STATS + 256;
static constexpr size_t O_RED   = O_LOG   + NB;
static constexpr size_t O_WBF   = O_RED   + 64;                 // bf16 region (u16)

// bf16 weight sub-offsets (u16 elements)
static constexpr size_t WB_MSG  = 0;
static constexpr size_t WB_SELF = WB_MSG  + 128 * 128;
static constexpr size_t WB_WIH0 = WB_SELF + 128 * 128;
static constexpr size_t WB_WHH0 = WB_WIH0 + 384 * 128;
static constexpr size_t WB_WIH1 = WB_WHH0 + 384 * 128;
static constexpr size_t WB_WHH1 = WB_WIH1 + 384 * 128;
static constexpr size_t WB_CWIH = WB_WHH1 + 384 * 128;
static constexpr size_t WB_CWHH = WB_CWIH + 384 * 128;
static constexpr size_t WB_LIN2 = WB_CWHH + 384 * 128;          // 128 x 256
static constexpr size_t WB_FST  = WB_LIN2 + 128 * 256;
static constexpr size_t WB_H1   = WB_FST  + 128 * 128;

extern "C" void kernel_launch(void* const* d_in, const int* in_sizes, int n_in,
                              void* d_out, int out_size, void* d_ws, size_t ws_size,
                              hipStream_t stream) {
    const float* X        = (const float*)d_in[0];
    const float* Y        = (const float*)d_in[1];
    const float* msg_W    = (const float*)d_in[2];
    const float* msg_b    = (const float*)d_in[3];
    const float* gWih0    = (const float*)d_in[4];
    const float* gWhh0    = (const float*)d_in[5];
    const float* gbih0    = (const float*)d_in[6];
    const float* gbhh0    = (const float*)d_in[7];
    const float* gWih1    = (const float*)d_in[8];
    const float* gWhh1    = (const float*)d_in[9];
    const float* gbih1    = (const float*)d_in[10];
    const float* gbhh1    = (const float*)d_in[11];
    const float* self_W   = (const float*)d_in[12];
    const float* self_b   = (const float*)d_in[13];
    const float* lin2_W   = (const float*)d_in[14];
    const float* lin2_b   = (const float*)d_in[15];
    const float* cWih     = (const float*)d_in[16];
    const float* cWhh     = (const float*)d_in[17];
    const float* cbih     = (const float*)d_in[18];
    const float* cbhh     = (const float*)d_in[19];
    const float* fst_W    = (const float*)d_in[20];
    const float* fst_b    = (const float*)d_in[21];
    const float* bn0_g    = (const float*)d_in[22];
    const float* bn0_b    = (const float*)d_in[23];
    const float* h1_W     = (const float*)d_in[24];
    const float* h1_b     = (const float*)d_in[25];
    const float* bn1_g    = (const float*)d_in[26];
    const float* bn1_b    = (const float*)d_in[27];
    const float* out_W    = (const float*)d_in[28];
    const float* out_b    = (const float*)d_in[29];

    float* ws  = (float*)d_ws;
    u16*   wbf = (u16*)(ws + O_WBF);
    float* XM  = ws + O_XMEAN;  float* TG  = ws + O_TGT;
    float* SA  = ws + O_SEQA;   float* SB  = ws + O_SEQB;
    float* HS  = ws + O_HSELF;  float* GI  = ws + O_GI;
    float* GH  = ws + O_GH;     float* H   = ws + O_H;
    float* CC  = ws + O_CC;     float* HX  = ws + O_HX;
    float* Z0  = ws + O_Z0;     float* Z1  = ws + O_Z1;
    float* ST  = ws + O_STATS;  float* LG  = ws + O_LOG;
    float* RD  = ws + O_RED;
    float* out = (float*)d_out;

    auto cvt = [&](const float* src, size_t dst, int n) {
        f2bf_kernel<<<(n + 255) / 256, 256, 0, stream>>>(src, wbf + dst, n);
    };
    auto gemm = [&](const float* A, size_t Wof, const float* bias, float* C,
                    int M, int N, int K, int act) {
        dim3 g(M / 64, N / 64);
        if (act)
            gemm_bias_kernel<1><<<g, 128, 0, stream>>>(A, wbf + Wof, bias, C, M, N, K);
        else
            gemm_bias_kernel<0><<<g, 128, 0, stream>>>(A, wbf + Wof, bias, C, M, N, K);
    };
    auto zero = [&](float* p, int n) {
        zero_kernel<<<(n + 255) / 256, 256, 0, stream>>>(p, n);
    };

    // ---- weight precision conversion (fp32 -> bf16, one-time per call, tiny) ----
    cvt(msg_W,  WB_MSG,  128 * 128);   cvt(self_W, WB_SELF, 128 * 128);
    cvt(gWih0,  WB_WIH0, 384 * 128);   cvt(gWhh0,  WB_WHH0, 384 * 128);
    cvt(gWih1,  WB_WIH1, 384 * 128);   cvt(gWhh1,  WB_WHH1, 384 * 128);
    cvt(cWih,   WB_CWIH, 384 * 128);   cvt(cWhh,   WB_CWHH, 384 * 128);
    cvt(lin2_W, WB_LIN2, 128 * 256);   cvt(fst_W,  WB_FST,  128 * 128);
    cvt(h1_W,   WB_H1,   128 * 128);

    // ---- Phase A: 484 MB stream; fold mean over neighbors BEFORE the msg matmul ----
    mean_target_kernel<<<NW / 8, 256, 0, stream>>>(X, XM, TG);

    // msg / self projections as flat (w*n) x 128 GEMMs
    gemm(XM, WB_MSG,  msg_b,  SA, NW, 128, 128, 0);   // seq input  [w][n][128]
    gemm(TG, WB_SELF, self_b, HS, NW, 128, 128, 0);   // h_self_t   [w][n][128]

    // ---- two GRU layers: hoist gi = x@Wih^T across the whole sequence ----
    const size_t GIW = (size_t)NB * 384, SQW = (size_t)NB * 128;
    {   // layer 0: SA -> SB
        gemm(SA, WB_WIH0, gbih0, GI, NW, 384, 128, 0);
        zero(H, NB * 128);
        for (int w = 0; w < WW; ++w) {
            gemm(H, WB_WHH0, gbhh0, GH, NB, 384, 128, 0);
            gru_combine_kernel<<<NB * 128 / 256, 256, 0, stream>>>(
                GI + (size_t)w * GIW, GH, H, H, SB + (size_t)w * SQW);
        }
    }
    {   // layer 1: SB -> SA   (SA now holds h_nei_t)
        gemm(SB, WB_WIH1, gbih1, GI, NW, 384, 128, 0);
        zero(H, NB * 128);
        for (int w = 0; w < WW; ++w) {
            gemm(H, WB_WHH1, gbhh1, GH, NB, 384, 128, 0);
            gru_combine_kernel<<<NB * 128 / 256, 256, 0, stream>>>(
                GI + (size_t)w * GIW, GH, H, H, SA + (size_t)w * SQW);
        }
    }

    // ---- fusion cell recurrence (7 steps); cell gi hoisted from h_self ----
    gemm(HS, WB_CWIH, cbih, GI, NW, 384, 128, 0);
    zero(HX, NB * 128);
    for (int w = 0; w < WW; ++w) {
        concat_kernel<<<NB * 256 / 256, 256, 0, stream>>>(HX, SA + (size_t)w * SQW, CC);
        gemm(CC, WB_LIN2, lin2_b, Z0, NB, 128, 256, 1);          // tanh fused -> hx2
        gemm(Z0, WB_CWHH, cbhh, GH, NB, 384, 128, 0);            // gh = hx2 @ Whh^T
        gru_combine_kernel<<<NB * 128 / 256, 256, 0, stream>>>(
            GI + (size_t)w * GIW, GH, Z0 /* h = hx2 */, HX, nullptr);
    }

    // ---- MLP head with training-mode batchnorm ----
    gemm(HX, WB_FST, fst_b, Z0, NB, 128, 128, 0);
    bn_stats_kernel<<<128, 256, 0, stream>>>(Z0, ST);
    bn_apply_relu_kernel<<<NB * 128 / 256, 256, 0, stream>>>(Z0, ST, bn0_g, bn0_b, Z1);
    gemm(Z1, WB_H1, h1_b, Z0, NB, 128, 128, 0);
    bn_stats_kernel<<<128, 256, 0, stream>>>(Z0, ST);
    bn_apply_relu_kernel<<<NB * 128 / 256, 256, 0, stream>>>(Z0, ST, bn1_g, bn1_b, Z1);

    // ---- logits, loss, sigmoid outputs ----
    logits_kernel<<<NB / 8, 256, 0, stream>>>(Z1, out_W, out_b, LG);
    loss_kernel<<<16, 256, 0, stream>>>(LG, Y, out, RD);
    loss_final_kernel<<<1, 32, 0, stream>>>(RD, out);
}
